// RNAPocketEncoderV3_13657996002052
// MI455X (gfx1250) — compile-verified, matrix operand order checked by reading, not run on hardware
//
#include <hip/hip_runtime.h>
#include <math.h>

typedef __attribute__((ext_vector_type(16))) _Float16 v16h;
typedef __attribute__((ext_vector_type(8)))  float    v8f;

__device__ __forceinline__ float fast_rcp(float x) { return __builtin_amdgcn_rcpf(x); }
__device__ __forceinline__ float sigm_f(float x) { return fast_rcp(1.f + __expf(-x)); }
__device__ __forceinline__ float silu_f(float x) { return x * sigm_f(x); }

__device__ __forceinline__ void atomicMaxFloat(float* addr, float val) {
  unsigned int* ua = (unsigned int*)addr;
  unsigned int old = *ua;
  while (true) {
    float f = __uint_as_float(old);
    if (f >= val) break;
    unsigned int assumed = old;
    old = atomicCAS(ua, assumed, __float_as_uint(val));
    if (old == assumed) break;
  }
}

// ---------------- WMMA helpers (gfx1250 wave32, 16x16x32 f16 -> f32) ----------------
// A fragment: lane m = lane&15, half = lane>>4; halves e<8 -> K=half*8+e, e>=8 -> K=16+half*8+(e-8)
// C/D: VGPR r -> M = r + 8*half, N = lane&15
__device__ __forceinline__ v8f wmma_tile_f16(const _Float16* __restrict__ A, int lda,
                                             const _Float16* __restrict__ B, int ldb,
                                             int K, int m0, int n0)
{
  const int lane = threadIdx.x & 31;
  const int half = lane >> 4;
  const int mn   = lane & 15;
  v8f acc = {};
  for (int k0 = 0; k0 < K; k0 += 32) {
    v16h a, b;
#pragma unroll
    for (int e = 0; e < 16; ++e) {
      const int k = k0 + ((e < 8) ? (half * 8 + e) : (8 + half * 8 + e));
      a[e] = A[(m0 + mn) * lda + k];
      b[e] = B[k * ldb + (n0 + mn)];
    }
    acc = __builtin_amdgcn_wmma_f32_16x16x32_f16(false, a, false, b, (short)0, acc, false, false);
  }
  return acc;
}

__device__ __forceinline__ void wmma_store_f16(v8f acc, _Float16* __restrict__ C, int ldc,
                                               int m0, int n0, const float* __restrict__ bias,
                                               bool do_silu)
{
  const int lane = threadIdx.x & 31;
  const int half = lane >> 4;
  const int n = n0 + (lane & 15);
  const float bv = bias[n];
#pragma unroll
  for (int r = 0; r < 8; ++r) {
    float x = acc[r] + bv;
    if (do_silu) x = silu_f(x);
    C[(m0 + r + 8 * half) * ldc + n] = (_Float16)x;
  }
}

// ---------------- Embedding: x[N,3] -> s[N,32] ----------------
__global__ __launch_bounds__(256) void embed_kernel(const float* __restrict__ x,
                                                    const float* __restrict__ P,
                                                    float* __restrict__ s, int N)
{
  __shared__ float W1[192], W2[4096], W3[2048], b1[64], b2[64], b3[32];
  const int tid = threadIdx.x;
  for (int i = tid; i < 192;  i += 256) W1[i] = P[i];
  for (int i = tid; i < 4096; i += 256) W2[i] = P[192 + i];
  for (int i = tid; i < 2048; i += 256) W3[i] = P[4288 + i];
  for (int i = tid; i < 64;   i += 256) b1[i] = P[6336 + i];
  for (int i = tid; i < 64;   i += 256) b2[i] = P[6400 + i];
  for (int i = tid; i < 32;   i += 256) b3[i] = P[6464 + i];
  __syncthreads();
  const int n = blockIdx.x * 256 + tid;
  if (n >= N) return;
  const float x0 = x[n*3], x1 = x[n*3+1], x2 = x[n*3+2];
  float h1[64], h2[64];
  for (int o = 0; o < 64; ++o)
    h1[o] = silu_f(b1[o] + x0*W1[o] + x1*W1[64+o] + x2*W1[128+o]);
  for (int o = 0; o < 64; ++o) {
    float a = b2[o];
    for (int k = 0; k < 64; ++k) a += h1[k]*W2[k*64+o];
    h2[o] = silu_f(a);
  }
  for (int o = 0; o < 32; ++o) {
    float a = b3[o];
    for (int k = 0; k < 64; ++k) a += h2[k]*W3[k*32+o];
    s[n*32+o] = a;
  }
}

// ---------------- E3 edge layer (bonded EA=2, nonbonded EA=3) ----------------
// e3p sorted offsets: A_s 0, B_v 1024, C_sv 1536, D_vv 2048, E_st 2304, F_tt 2560,
// G_t 2624, G_v 2880, SC_s 3392, SC_t 4416, SC_v 4480, W1 4736, W2, b1, b2
template<int EA>
__global__ __launch_bounds__(256) void e3_edge_kernel(
    const float* __restrict__ pos, const float* __restrict__ eattr,
    const int* __restrict__ eidx,
    const float* __restrict__ s, const float* __restrict__ v, const float* __restrict__ t,
    const float* __restrict__ P,
    float* __restrict__ aggS, float* __restrict__ aggV, float* __restrict__ aggT,
    int E, float inv_denom)
{
  constexpr int KIN = 8 + EA;
  constexpr int W1O = 4736;
  constexpr int W2O = W1O + KIN * 64;
  constexpr int B1O = W2O + 64 * 112;
  constexpr int B2O = B1O + 64;

  __shared__ _Float16 sFeat[64*32];
  __shared__ _Float16 sW1[32*64];
  __shared__ _Float16 sHid[64*64];
  __shared__ _Float16 sW2[64*112];
  __shared__ _Float16 sR[64*112];
  __shared__ _Float16 sEW[2624];     // A_s..F_tt contiguous
  __shared__ float    sB1[64], sB2[112];
  __shared__ _Float16 sS[64*32];
  __shared__ float    sVdot[64*16];
  __shared__ float    sU[64*3], sY2[64*5];
  __shared__ int      sSrc[64], sDst[64];

  const int tid = threadIdx.x;
  const int e0 = blockIdx.x * 64;
  // Prefetch next tile's index/attr rows (global_prefetch_b8; speculative, OOB-safe)
  if (tid == 0) {
    __builtin_prefetch(&eidx[e0 + 64], 0, 0);
    __builtin_prefetch(&eidx[E + e0 + 64], 0, 0);
    __builtin_prefetch(&eattr[(size_t)(e0 + 64) * EA], 0, 0);
  }

  for (int i = tid; i < 32*64; i += 256) {
    int r = i >> 6;
    sW1[i] = (r < KIN) ? (_Float16)P[W1O + i] : (_Float16)0.f;
  }
  for (int i = tid; i < 64*112; i += 256) sW2[i] = (_Float16)P[W2O + i];
  for (int i = tid; i < 2624;   i += 256) sEW[i] = (_Float16)P[i];
  for (int i = tid; i < 64;     i += 256) sB1[i] = P[B1O + i];
  for (int i = tid; i < 112;    i += 256) sB2[i] = P[B2O + i];

  if (tid < 64) {
    const int e = e0 + tid;
    if (e < E) {
      const int src = eidx[e], dst = eidx[E + e];
      sSrc[tid] = src; sDst[tid] = dst;
      const float rx = pos[src*3]   - pos[dst*3];
      const float ry = pos[src*3+1] - pos[dst*3+1];
      const float rz = pos[src*3+2] - pos[dst*3+2];
      const float d  = sqrtf(rx*rx + ry*ry + rz*rz);
      const float iv = fast_rcp(d + 1e-8f);
      const float ux = rx*iv, uy = ry*iv, uz = rz*iv;
      sU[tid*3] = ux; sU[tid*3+1] = uy; sU[tid*3+2] = uz;
      const float s3 = 1.7320508075688772f;
      sY2[tid*5+0] = s3*ux*uy; sY2[tid*5+1] = s3*uy*uz;
      sY2[tid*5+2] = 0.5f*(3.f*uz*uz - 1.f);
      sY2[tid*5+3] = s3*ux*uz; sY2[tid*5+4] = 0.5f*s3*(ux*ux - uy*uy);
      const float gamma = (8.f/6.f)*(8.f/6.f);
      for (int j = 0; j < 8; ++j) {
        float c = 6.f * (float)j / 7.f, df = d - c;
        sFeat[tid*32+j] = (_Float16)__expf(-gamma*df*df);
      }
      for (int j = 0; j < EA; ++j) sFeat[tid*32+8+j] = (_Float16)eattr[e*EA+j];
      for (int j = KIN; j < 32; ++j) sFeat[tid*32+j] = (_Float16)0.f;
      for (int c = 0; c < 32; ++c) sS[tid*32+c] = (_Float16)s[src*32+c];
      for (int c = 0; c < 16; ++c) {
        const float* vp = &v[src*48 + c*3];
        sVdot[tid*16+c] = vp[0]*ux + vp[1]*uy + vp[2]*uz;
      }
    } else {
      sSrc[tid] = -1; sDst[tid] = -1;
      for (int j = 0; j < 32; ++j) { sFeat[tid*32+j] = (_Float16)0.f; sS[tid*32+j] = (_Float16)0.f; }
      for (int c = 0; c < 16; ++c) sVdot[tid*16+c] = 0.f;
      for (int xk = 0; xk < 3; ++xk) sU[tid*3+xk] = 0.f;
      for (int xk = 0; xk < 5; ++xk) sY2[tid*5+xk] = 0.f;
    }
  }
  __syncthreads();

  const int wave = tid >> 5;
  // GEMM1: [64 x 32] x [32 x 64] -> hid (silu), 16 tiles over 8 waves
  for (int ti = wave; ti < 16; ti += 8) {
    const int mt = ti >> 2, nt = ti & 3;
    v8f acc = wmma_tile_f16(sFeat, 32, sW1, 64, 32, mt*16, nt*16);
    wmma_store_f16(acc, sHid, 64, mt*16, nt*16, sB1, true);
  }
  __syncthreads();
  // GEMM2: [64 x 64] x [64 x 112] -> R, 28 tiles
  for (int ti = wave; ti < 28; ti += 8) {
    const int mt = ti / 7, nt = ti % 7;
    v8f acc = wmma_tile_f16(sHid, 64, sW2, 112, 64, mt*16, nt*16);
    wmma_store_f16(acc, sR, 112, mt*16, nt*16, sB2, false);
  }
  __syncthreads();

  // Epilogue: 4 threads per edge
  const int e   = tid >> 2;
  const int sub = tid & 3;
  const int dst = sDst[e];
  if (dst >= 0) {
    const int src = sSrc[e];
    const _Float16* R = &sR[e*112];
    // m_s
    for (int c = sub; c < 32; c += 4) {
      float aA = 0.f, aB = 0.f;
      for (int k = 0; k < 32; ++k) aA += (float)sS[e*32+k] * (float)sEW[k*32 + c];
      for (int k = 0; k < 16; ++k) aB += sVdot[e*16+k] * (float)sEW[1024 + k*32 + c];
      const float ms = (float)R[c]*aA + (float)R[32+c]*aB;
      atomicAdd(&aggS[dst*32 + c], ms * inv_denom);
    }
    // m_v
    for (int d = sub; d < 16; d += 4) {
      float cs = 0.f;
      for (int k = 0; k < 32; ++k) cs += (float)sS[e*32+k] * (float)sEW[1536 + k*16 + d];
      const float rsv = (float)R[64+d], rvv = (float)R[80+d];
      float dv[3] = {0.f, 0.f, 0.f};
      for (int c = 0; c < 16; ++c) {
        const float w = (float)sEW[2048 + c*16 + d];
        const float* vp = &v[src*48 + c*3];
        dv[0] += vp[0]*w; dv[1] += vp[1]*w; dv[2] += vp[2]*w;
      }
      for (int xk = 0; xk < 3; ++xk) {
        const float mv = rsv*cs*sU[e*3+xk] + rvv*dv[xk];
        atomicAdd(&aggV[dst*48 + d*3 + xk], mv * inv_denom);
      }
    }
    // m_t
    for (int d = sub; d < 8; d += 4) {
      float es = 0.f;
      for (int k = 0; k < 32; ++k) es += (float)sS[e*32+k] * (float)sEW[2304 + k*8 + d];
      const float rst = (float)R[96+d], rtt = (float)R[104+d];
      float dt[5] = {0.f,0.f,0.f,0.f,0.f};
      for (int c = 0; c < 8; ++c) {
        const float w = (float)sEW[2560 + c*8 + d];
        const float* tp = &t[src*40 + c*5];
        for (int xk = 0; xk < 5; ++xk) dt[xk] += tp[xk]*w;
      }
      for (int xk = 0; xk < 5; ++xk) {
        const float mt_ = rst*es*sY2[e*5+xk] + rtt*dt[xk];
        atomicAdd(&aggT[dst*40 + d*5 + xk], mt_ * inv_denom);
      }
    }
  }
}

// ---------------- Angle (mode 0) / Dihedral (mode 1) layers ----------------
// angp sorted offsets: G_t 0, G_v 256, W1 768, W2 768+din*64, b1, b2, ln_b, ln_g
__global__ __launch_bounds__(256) void angle_kernel(
    const float* __restrict__ pos, const float* __restrict__ attr,
    const int* __restrict__ idx, const float* __restrict__ s,
    const float* __restrict__ P, float* __restrict__ agg,
    int T, int adim, int mode, float inv_denom)
{
  __shared__ _Float16 sInp[64*96];
  __shared__ _Float16 sW1[96*64];
  __shared__ _Float16 sHid[64*64];
  __shared__ _Float16 sW2[64*32];
  __shared__ float sB1[64], sB2[32];
  __shared__ int sJ[64];

  const int tid = threadIdx.x;
  const int din = 66 + adim;
  const int W1O = 768;
  const int W2O = 768 + din*64;
  const int B1O = W2O + 2048;
  const int B2O = B1O + 64;

  const int e0 = blockIdx.x * 64;
  if (tid == 0) {
    __builtin_prefetch(&idx[e0 + 64], 0, 0);
    __builtin_prefetch(&idx[T + e0 + 64], 0, 0);
    __builtin_prefetch(&attr[(size_t)(e0 + 64) * adim], 0, 0);
  }

  for (int i = tid; i < 96*64; i += 256) {
    int r = i >> 6;
    sW1[i] = (r < din) ? (_Float16)P[W1O + i] : (_Float16)0.f;
  }
  for (int i = tid; i < 64*32; i += 256) sW2[i] = (_Float16)P[W2O + i];
  for (int i = tid; i < 64; i += 256) sB1[i] = P[B1O + i];
  for (int i = tid; i < 32; i += 256) sB2[i] = P[B2O + i];

  if (tid < 64) {
    const int e = e0 + tid;
    _Float16* row = &sInp[tid*96];
    if (e < T) {
      float cosv, sinv; int a, b, j;
      if (mode == 0) {
        const int i_ = idx[e], jj = idx[T+e], k_ = idx[2*T+e];
        j = jj; a = i_; b = k_;
        float r1[3], r2[3];
        for (int xk = 0; xk < 3; ++xk) { r1[xk] = pos[i_*3+xk]-pos[jj*3+xk]; r2[xk] = pos[k_*3+xk]-pos[jj*3+xk]; }
        const float n1 = sqrtf(r1[0]*r1[0]+r1[1]*r1[1]+r1[2]*r1[2]);
        const float n2 = sqrtf(r2[0]*r2[0]+r2[1]*r2[1]+r2[2]*r2[2]);
        cosv = (r1[0]*r2[0]+r1[1]*r2[1]+r1[2]*r2[2]) * fast_rcp(n1*n2 + 1e-8f);
        float s2 = 1.f - cosv*cosv;
        s2 = s2 < 0.f ? 0.f : (s2 > 1.f ? 1.f : s2);
        sinv = sqrtf(s2);
      } else {
        const int i_ = idx[e], jj = idx[T+e], k_ = idx[2*T+e], l_ = idx[3*T+e];
        j = jj; a = jj; b = k_;
        float b1v[3], b2v[3], b3v[3];
        for (int xk = 0; xk < 3; ++xk) {
          b1v[xk] = pos[jj*3+xk]-pos[i_*3+xk];
          b2v[xk] = pos[k_*3+xk]-pos[jj*3+xk];
          b3v[xk] = pos[l_*3+xk]-pos[k_*3+xk];
        }
        float n1[3] = { b1v[1]*b2v[2]-b1v[2]*b2v[1], b1v[2]*b2v[0]-b1v[0]*b2v[2], b1v[0]*b2v[1]-b1v[1]*b2v[0] };
        float n2[3] = { b2v[1]*b3v[2]-b2v[2]*b3v[1], b2v[2]*b3v[0]-b2v[0]*b3v[2], b2v[0]*b3v[1]-b2v[1]*b3v[0] };
        const float nn1 = sqrtf(n1[0]*n1[0]+n1[1]*n1[1]+n1[2]*n1[2]);
        const float nn2 = sqrtf(n2[0]*n2[0]+n2[1]*n2[1]+n2[2]*n2[2]);
        const float nb2 = sqrtf(b2v[0]*b2v[0]+b2v[1]*b2v[1]+b2v[2]*b2v[2]);
        cosv = (n1[0]*n2[0]+n1[1]*n2[1]+n1[2]*n2[2]) * fast_rcp(nn1*nn2 + 1e-8f);
        float c3[3] = { n1[1]*n2[2]-n1[2]*n2[1], n1[2]*n2[0]-n1[0]*n2[2], n1[0]*n2[1]-n1[1]*n2[0] };
        sinv = (c3[0]*b2v[0]+c3[1]*b2v[1]+c3[2]*b2v[2]) * fast_rcp(nn1*nn2*nb2 + 1e-8f);
      }
      sJ[tid] = j;
      for (int c = 0; c < 32; ++c) row[c]    = (_Float16)s[a*32+c];
      for (int c = 0; c < 32; ++c) row[32+c] = (_Float16)s[b*32+c];
      for (int c = 0; c < adim; ++c) row[64+c] = (_Float16)attr[e*adim+c];
      row[64+adim] = (_Float16)cosv; row[65+adim] = (_Float16)sinv;
      for (int c = din; c < 96; ++c) row[c] = (_Float16)0.f;
    } else {
      sJ[tid] = -1;
      for (int c = 0; c < 96; ++c) row[c] = (_Float16)0.f;
    }
  }
  __syncthreads();

  const int wave = tid >> 5;
  for (int ti = wave; ti < 16; ti += 8) {
    const int mt = ti >> 2, nt = ti & 3;
    v8f acc = wmma_tile_f16(sInp, 96, sW1, 64, 96, mt*16, nt*16);
    wmma_store_f16(acc, sHid, 64, mt*16, nt*16, sB1, true);
  }
  __syncthreads();
  // GEMM2: [64x64] x [64x32] -> scatter to agg[j]
  {
    const int mt = wave >> 1, nt = wave & 1;
    v8f acc = wmma_tile_f16(sHid, 64, sW2, 32, 64, mt*16, nt*16);
    const int lane = tid & 31;
    const int half = lane >> 4;
    const int n = nt*16 + (lane & 15);
    const float bv = sB2[n];
#pragma unroll
    for (int r = 0; r < 8; ++r) {
      const int m = mt*16 + r + 8*half;
      const int j = sJ[m];
      if (j >= 0) atomicAdd(&agg[j*32 + n], (acc[r] + bv) * inv_denom);
    }
  }
}

// ---------------- Per-node combine + EquivariantLayerNorm ----------------
__global__ __launch_bounds__(256) void combine_kernel(
    const float* __restrict__ s, const float* __restrict__ v, const float* __restrict__ t,
    const float* __restrict__ aggSb, const float* __restrict__ aggVb, const float* __restrict__ aggTb,
    const float* __restrict__ aggA, const float* __restrict__ aggD,
    const float* __restrict__ aggSn, const float* __restrict__ aggVn, const float* __restrict__ aggTn,
    const float* __restrict__ Pb, const float* __restrict__ Pn,
    const float* __restrict__ Pa, const float* __restrict__ Pd,
    const float* __restrict__ Pagg,
    const float* __restrict__ wA, const float* __restrict__ wD, const float* __restrict__ wN,
    int lnA_off, int lnD_off,
    float* __restrict__ so, float* __restrict__ vo, float* __restrict__ to_, int N)
{
  __shared__ float SCs[1024], SCv[256], SCt[64], GvB[512], GtB[256];
  __shared__ float GvN[512], GtN[256];
  __shared__ float GvA[512], GtA[256], lnA[64];
  __shared__ float GvD[512], GtD[256], lnD[64];
  __shared__ float lnG[64];
  const int tid = threadIdx.x;
  for (int i = tid; i < 1024; i += 256) SCs[i] = Pb[3392 + i];
  for (int i = tid; i < 256;  i += 256) SCv[i] = Pb[4480 + i];
  for (int i = tid; i < 64;   i += 256) SCt[i] = Pb[4416 + i];
  for (int i = tid; i < 512;  i += 256) GvB[i] = Pb[2880 + i];
  for (int i = tid; i < 256;  i += 256) GtB[i] = Pb[2624 + i];
  for (int i = tid; i < 512;  i += 256) GvN[i] = Pn[2880 + i];
  for (int i = tid; i < 256;  i += 256) GtN[i] = Pn[2624 + i];
  for (int i = tid; i < 512;  i += 256) GvA[i] = Pa[256 + i];
  for (int i = tid; i < 256;  i += 256) GtA[i] = Pa[i];
  for (int i = tid; i < 64;   i += 256) lnA[i] = Pa[lnA_off + i];   // [ln_b(32), ln_g(32)]
  for (int i = tid; i < 512;  i += 256) GvD[i] = Pd[256 + i];
  for (int i = tid; i < 256;  i += 256) GtD[i] = Pd[i];
  for (int i = tid; i < 64;   i += 256) lnD[i] = Pd[lnD_off + i];
  for (int i = tid; i < 64;   i += 256) lnG[i] = Pagg[i];           // [b(32), g(32)]
  __syncthreads();

  const int n = blockIdx.x * 256 + tid;
  if (n >= N) return;
  const float aw = wA[0], dw = wD[0], nw = wN[0];

  float sv[32], vv[48], tv[40];
  for (int c = 0; c < 32; ++c) sv[c] = s[n*32+c];
  for (int c = 0; c < 48; ++c) vv[c] = v[n*48+c];
  for (int c = 0; c < 40; ++c) tv[c] = t[n*40+c];

  // bonded (use_sc + resnet)
  float as[32], snew[32], vnew[48], tnew[40];
  for (int c = 0; c < 32; ++c) {
    float a = aggSb[n*32+c];
    for (int k = 0; k < 32; ++k) a += sv[k]*SCs[k*32+c];
    as[c] = a;
    snew[c] = sv[c] + silu_f(a);
  }
  for (int d = 0; d < 16; ++d) {
    float g = 0.f;
    for (int k = 0; k < 32; ++k) g += as[k]*GvB[k*16+d];
    g = sigm_f(g);
    for (int xk = 0; xk < 3; ++xk) {
      float a = aggVb[n*48 + d*3 + xk];
      for (int c = 0; c < 16; ++c) a += vv[c*3+xk]*SCv[c*16+d];
      vnew[d*3+xk] = vv[d*3+xk] + a*g;
    }
  }
  for (int d = 0; d < 8; ++d) {
    float g = 0.f;
    for (int k = 0; k < 32; ++k) g += as[k]*GtB[k*8+d];
    g = sigm_f(g);
    for (int xk = 0; xk < 5; ++xk) {
      float a = aggTb[n*40 + d*5 + xk];
      for (int c = 0; c < 8; ++c) a += tv[c*5+xk]*SCt[c*8+d];
      tnew[d*5+xk] = tv[d*5+xk] + a*g;
    }
  }
  // angle
  {
    float an[32], mu = 0.f;
    for (int c = 0; c < 32; ++c) { an[c] = aggA[n*32+c]; mu += an[c]; }
    mu /= 32.f;
    float var = 0.f;
    for (int c = 0; c < 32; ++c) { float df = an[c]-mu; var += df*df; }
    const float rs = rsqrtf(var/32.f + 1e-5f);
    for (int c = 0; c < 32; ++c) snew[c] += aw * ((an[c]-mu)*rs*lnA[32+c] + lnA[c]);
    for (int d = 0; d < 16; ++d) {
      float g = 0.f;
      for (int k = 0; k < 32; ++k) g += an[k]*GvA[k*16+d];
      g = sigm_f(g);
      for (int xk = 0; xk < 3; ++xk) vnew[d*3+xk] += aw * vv[d*3+xk] * g;
    }
    for (int d = 0; d < 8; ++d) {
      float g = 0.f;
      for (int k = 0; k < 32; ++k) g += an[k]*GtA[k*8+d];
      g = sigm_f(g);
      for (int xk = 0; xk < 5; ++xk) tnew[d*5+xk] += aw * tv[d*5+xk] * g;
    }
  }
  // dihedral
  {
    float an[32], mu = 0.f;
    for (int c = 0; c < 32; ++c) { an[c] = aggD[n*32+c]; mu += an[c]; }
    mu /= 32.f;
    float var = 0.f;
    for (int c = 0; c < 32; ++c) { float df = an[c]-mu; var += df*df; }
    const float rs = rsqrtf(var/32.f + 1e-5f);
    for (int c = 0; c < 32; ++c) snew[c] += dw * ((an[c]-mu)*rs*lnD[32+c] + lnD[c]);
    for (int d = 0; d < 16; ++d) {
      float g = 0.f;
      for (int k = 0; k < 32; ++k) g += an[k]*GvD[k*16+d];
      g = sigm_f(g);
      for (int xk = 0; xk < 3; ++xk) vnew[d*3+xk] += dw * vv[d*3+xk] * g;
    }
    for (int d = 0; d < 8; ++d) {
      float g = 0.f;
      for (int k = 0; k < 32; ++k) g += an[k]*GtD[k*8+d];
      g = sigm_f(g);
      for (int xk = 0; xk < 5; ++xk) tnew[d*5+xk] += dw * tv[d*5+xk] * g;
    }
  }
  // nonbonded (no sc, no resnet)
  {
    float ns[32];
    for (int c = 0; c < 32; ++c) { ns[c] = aggSn[n*32+c]; snew[c] += nw * silu_f(ns[c]); }
    for (int d = 0; d < 16; ++d) {
      float g = 0.f;
      for (int k = 0; k < 32; ++k) g += ns[k]*GvN[k*16+d];
      g = sigm_f(g);
      for (int xk = 0; xk < 3; ++xk) vnew[d*3+xk] += nw * aggVn[n*48+d*3+xk] * g;
    }
    for (int d = 0; d < 8; ++d) {
      float g = 0.f;
      for (int k = 0; k < 32; ++k) g += ns[k]*GtN[k*8+d];
      g = sigm_f(g);
      for (int xk = 0; xk < 5; ++xk) tnew[d*5+xk] += nw * aggTn[n*40+d*5+xk] * g;
    }
  }
  // aggregate LayerNorm on scalars
  {
    float mu = 0.f;
    for (int c = 0; c < 32; ++c) mu += snew[c];
    mu /= 32.f;
    float var = 0.f;
    for (int c = 0; c < 32; ++c) { float df = snew[c]-mu; var += df*df; }
    const float rs = rsqrtf(var/32.f + 1e-5f);
    for (int c = 0; c < 32; ++c) so[n*32+c] = (snew[c]-mu)*rs*lnG[32+c] + lnG[c];
  }
  for (int c = 0; c < 48; ++c) vo[n*48+c] = vnew[c];
  for (int c = 0; c < 40; ++c) to_[n*40+c] = tnew[c];
}

// ---------------- Pooling ----------------
__global__ __launch_bounds__(256) void inv_scores_kernel(
    const float* __restrict__ s, const float* __restrict__ v, const float* __restrict__ t,
    const float* __restrict__ P, float* __restrict__ inv, float* __restrict__ sc, int N)
{
  __shared__ float W1[56*128], b1[128], W2[512], b2[4];
  const int tid = threadIdx.x;
  for (int i = tid; i < 56*128; i += 256) W1[i] = P[i];
  for (int i = tid; i < 512;    i += 256) W2[i] = P[7168 + i];
  for (int i = tid; i < 128;    i += 256) b1[i] = P[7680 + i];
  for (int i = tid; i < 4;      i += 256) b2[i] = P[7808 + i];
  __syncthreads();
  const int n = blockIdx.x * 256 + tid;
  if (n >= N) return;
  float iv[56];
  for (int c = 0; c < 32; ++c) iv[c] = s[n*32+c];
  for (int c = 0; c < 16; ++c) {
    const float* vp = &v[n*48 + c*3];
    iv[32+c] = sqrtf(vp[0]*vp[0]+vp[1]*vp[1]+vp[2]*vp[2]);
  }
  for (int c = 0; c < 8; ++c) {
    const float* tp = &t[n*40 + c*5];
    float a = 0.f;
    for (int xk = 0; xk < 5; ++xk) a += tp[xk]*tp[xk];
    iv[48+c] = sqrtf(a);
  }
  for (int c = 0; c < 56; ++c) inv[n*56+c] = iv[c];
  float h[128];
  for (int o = 0; o < 128; ++o) {
    float a = b1[o];
    for (int k = 0; k < 56; ++k) a += iv[k]*W1[k*128+o];
    h[o] = silu_f(a);
  }
  for (int o = 0; o < 4; ++o) {
    float a = b2[o];
    for (int k = 0; k < 128; ++k) a += h[k]*W2[k*4+o];
    sc[n*4+o] = a;
  }
}

__global__ __launch_bounds__(256) void init_max_kernel(float* g, int n) {
  const int i = blockIdx.x*256 + threadIdx.x;
  if (i < n) g[i] = -3e38f;
}
__global__ __launch_bounds__(256) void segmax_kernel(const float* __restrict__ sc,
                                                     const int* __restrict__ batch,
                                                     float* __restrict__ gmax, int N) {
  const int n = blockIdx.x*256 + threadIdx.x;
  if (n >= N) return;
  const int g = batch[n];
  for (int h = 0; h < 4; ++h) atomicMaxFloat(&gmax[g*4+h], sc[n*4+h]);
}
__global__ __launch_bounds__(256) void expsum_kernel(const float* __restrict__ sc,
                                                     const int* __restrict__ batch,
                                                     const float* __restrict__ gmax,
                                                     float* __restrict__ eb, float* __restrict__ z, int N) {
  const int n = blockIdx.x*256 + threadIdx.x;
  if (n >= N) return;
  const int g = batch[n];
  for (int h = 0; h < 4; ++h) {
    const float e = __expf(sc[n*4+h] - gmax[g*4+h]);
    eb[n*4+h] = e;
    atomicAdd(&z[g*4+h], e);
  }
}
__global__ __launch_bounds__(256) void pool_kernel(const float* __restrict__ inv,
                                                   const float* __restrict__ eb,
                                                   const float* __restrict__ z,
                                                   const int* __restrict__ batch,
                                                   float* __restrict__ pooled, int N) {
  const int n = blockIdx.x*256 + threadIdx.x;
  if (n >= N) return;
  const int g = batch[n];
  for (int h = 0; h < 4; ++h) {
    const float a = eb[n*4+h] * fast_rcp(z[g*4+h] + 1e-9f);
    for (int d = 0; d < 14; ++d)
      atomicAdd(&pooled[g*56 + h*14 + d], a * inv[n*56 + h*14 + d]);
  }
}

__global__ __launch_bounds__(256) void out_kernel(const float* __restrict__ pooled,
                                                  const float* __restrict__ P,
                                                  float* __restrict__ out, int B) {
  __shared__ float sp[56], red1[256], red2[256];
  __shared__ float mu_s, rs_s;
  const int g = blockIdx.x, tid = threadIdx.x;
  if (tid < 56) sp[tid] = pooled[g*56 + tid];
  __syncthreads();
  float val[2];
  for (int h = 0; h < 2; ++h) {
    const int o = tid + h*256;
    float a = P[28672 + o];
    for (int k = 0; k < 56; ++k) a += sp[k]*P[k*512 + o];
    val[h] = a;
  }
  red1[tid] = val[0] + val[1];
  red2[tid] = val[0]*val[0] + val[1]*val[1];
  __syncthreads();
  for (int st = 128; st > 0; st >>= 1) {
    if (tid < st) { red1[tid] += red1[tid+st]; red2[tid] += red2[tid+st]; }
    __syncthreads();
  }
  if (tid == 0) {
    const float mu = red1[0]/512.f;
    mu_s = mu;
    rs_s = rsqrtf(red2[0]/512.f - mu*mu + 1e-5f);
  }
  __syncthreads();
  for (int h = 0; h < 2; ++h) {
    const int o = tid + h*256;
    out[g*512 + o] = (val[h]-mu_s)*rs_s*P[29696+o] + P[29184+o];
  }
}

// ---------------- Host orchestration ----------------
extern "C" void kernel_launch(void* const* d_in, const int* in_sizes, int n_in,
                              void* d_out, int out_size, void* d_ws, size_t ws_size,
                              hipStream_t stream) {
  (void)n_in; (void)ws_size;
  const float* x          = (const float*)d_in[0];
  const float* pos        = (const float*)d_in[1];
  const float* edge_attr  = (const float*)d_in[2];
  const float* tri_attr   = (const float*)d_in[3];
  const float* quad_attr  = (const float*)d_in[4];
  const float* nb_attr    = (const float*)d_in[5];
  const float* prm        = (const float*)d_in[6];
  const int*   eidx       = (const int*)d_in[7];
  const int*   tidx       = (const int*)d_in[8];
  const int*   qidx       = (const int*)d_in[9];
  const int*   nbidx      = (const int*)d_in[10];
  const int*   batch      = (const int*)d_in[11];

  const int N   = in_sizes[0] / 3;
  const int E   = in_sizes[2] / 2;
  const int T   = in_sizes[3] / 2;
  const int Q   = in_sizes[4] / 3;
  const int ENB = in_sizes[5] / 3;
  const int B   = out_size / 512;

  // Param offsets: jax tree_flatten sorted-key order.
  size_t off = 0;
  size_t aggln_off[4]; for (int i = 0; i < 4; ++i) { aggln_off[i] = off; off += 64; }
  size_t ang_off[4];   for (int i = 0; i < 4; ++i) { ang_off[i] = off; off += 7328; }
  const size_t aw_off = off; off += 1;
  size_t bon_off[4];   for (int i = 0; i < 4; ++i) { bon_off[i] = off; off += 12720; }
  size_t dih_off[4];   for (int i = 0; i < 4; ++i) { dih_off[i] = off; off += 7392; }
  const size_t dw_off = off; off += 1;
  const size_t emb_off = off; off += 6496;
  size_t nb_off[4];    for (int i = 0; i < 4; ++i) { nb_off[i] = off; off += 12784; }
  const size_t nw_off = off; off += 1;
  const size_t out_off = off; off += 30208;
  const size_t pool_off = off; off += 7812;

  // Workspace carve-up (floats)
  float* w = (float*)d_ws;
  auto alloc = [&](size_t nf) { float* p = w; w += nf; return p; };
  const size_t NN = (size_t)N;
  float* s0 = alloc(NN*32); float* v0 = alloc(NN*48); float* t0 = alloc(NN*40);
  float* s1 = alloc(NN*32); float* v1 = alloc(NN*48); float* t1 = alloc(NN*40);
  float* aggSb = alloc(NN*32); float* aggVb = alloc(NN*48); float* aggTb = alloc(NN*40);
  float* aggSn = alloc(NN*32); float* aggVn = alloc(NN*48); float* aggTn = alloc(NN*40);
  float* aggA = alloc(NN*32); float* aggD = alloc(NN*32);
  float* invb = alloc(NN*56); float* sc = alloc(NN*4); float* eb = alloc(NN*4);
  float* gmax = alloc((size_t)B*4); float* z = alloc((size_t)B*4); float* pooled = alloc((size_t)B*56);

  const int nbn = (N + 255) / 256;
  embed_kernel<<<nbn, 256, 0, stream>>>(x, prm + emb_off, s0, N);
  (void)hipMemsetAsync(v0, 0, NN*48*sizeof(float), stream);
  (void)hipMemsetAsync(t0, 0, NN*40*sizeof(float), stream);

  const float invd_b = 1.f / sqrtf((float)E  / (float)N);
  const float invd_a = 1.f / sqrtf((float)T  / (float)N);
  const float invd_d = 1.f / sqrtf((float)Q  / (float)N);
  const float invd_n = 1.f / sqrtf((float)ENB / (float)N);

  float* scur = s0; float* vcur = v0; float* tcur = t0;
  float* snxt = s1; float* vnxt = v1; float* tnxt = t1;
  const size_t aggBytes = (NN*(32+48+40)*2 + NN*64) * sizeof(float); // aggSb..aggD contiguous

  for (int li = 0; li < 4; ++li) {
    (void)hipMemsetAsync(aggSb, 0, aggBytes, stream);
    e3_edge_kernel<2><<<(E   + 63) / 64, 256, 0, stream>>>(pos, edge_attr, eidx, scur, vcur, tcur,
        prm + bon_off[li], aggSb, aggVb, aggTb, E, invd_b);
    angle_kernel<<<(T + 63) / 64, 256, 0, stream>>>(pos, tri_attr, tidx, scur,
        prm + ang_off[li], aggA, T, 2, 0, invd_a);
    angle_kernel<<<(Q + 63) / 64, 256, 0, stream>>>(pos, quad_attr, qidx, scur,
        prm + dih_off[li], aggD, Q, 3, 1, invd_d);
    e3_edge_kernel<3><<<(ENB + 63) / 64, 256, 0, stream>>>(pos, nb_attr, nbidx, scur, vcur, tcur,
        prm + nb_off[li], aggSn, aggVn, aggTn, ENB, invd_n);
    combine_kernel<<<nbn, 256, 0, stream>>>(scur, vcur, tcur,
        aggSb, aggVb, aggTb, aggA, aggD, aggSn, aggVn, aggTn,
        prm + bon_off[li], prm + nb_off[li], prm + ang_off[li], prm + dih_off[li],
        prm + aggln_off[li], prm + aw_off, prm + dw_off, prm + nw_off,
        7264 /* angle ln_b (din=68) */, 7328 /* dihedral ln_b (din=69) */,
        snxt, vnxt, tnxt, N);
    float* ts_;
    ts_ = scur; scur = snxt; snxt = ts_;
    ts_ = vcur; vcur = vnxt; vnxt = ts_;
    ts_ = tcur; tcur = tnxt; tnxt = ts_;
  }

  inv_scores_kernel<<<nbn, 256, 0, stream>>>(scur, vcur, tcur, prm + pool_off, invb, sc, N);
  init_max_kernel<<<(B*4 + 255) / 256, 256, 0, stream>>>(gmax, B*4);
  segmax_kernel<<<nbn, 256, 0, stream>>>(sc, batch, gmax, N);
  (void)hipMemsetAsync(z, 0, (size_t)B*4*sizeof(float), stream);
  (void)hipMemsetAsync(pooled, 0, (size_t)B*56*sizeof(float), stream);
  expsum_kernel<<<nbn, 256, 0, stream>>>(sc, batch, gmax, eb, z, N);
  pool_kernel<<<nbn, 256, 0, stream>>>(invb, eb, z, batch, pooled, N);
  out_kernel<<<B, 256, 0, stream>>>(pooled, prm + out_off, (float*)d_out, B);
}
